// Decoder_21758304321761
// MI455X (gfx1250) — compile-verified, matrix operand order checked by reading, not run on hardware
//
#include <hip/hip_runtime.h>
#include <hip/hip_bf16.h>

// ---------------------------------------------------------------------------
// Problem constants (from the reference)
// ---------------------------------------------------------------------------
constexpr int B  = 32;
constexpr int S  = 128;   // unused by the math (encoder_outputs never read)
constexpr int H  = 1024;
constexpr int V  = 32000;
constexpr int T  = 100;
constexpr int L  = 4;
constexpr int H3 = 3 * H;          // 3072
constexpr int K0 = 2 * H;          // 2048 (layer-0 GRU input width)
constexpr int NT_G  = H3 / 16;     // 192 N-tiles for GRU gates
constexpr int KT_H  = H / 32;      // 32 K-tiles for K=1024
constexpr int KT_X0 = K0 / 32;     // 64 K-tiles for K=2048
constexpr int NT_V  = V / 16;      // 2000 N-tiles for vocab projection
constexpr int FRAG  = 512;         // bf16 elements per 16x32 (A) / 32x16 (B) fragment

typedef __attribute__((ext_vector_type(16))) __bf16 v16bf;
typedef __attribute__((ext_vector_type(8)))  float  v8f;

// ---------------------------------------------------------------------------
// WMMA 16-bit operand lane layout (CDNA5 ISA §7.12.2):
//   lane = half*16 + (m or n);  VGPR r holds element pair e=2r,2r+1
//   K index: r<4 : k = 2r + 8*half ; r>=4 : k = 16 + 2*(r-4) + 8*half  (+ e&1)
// ---------------------------------------------------------------------------
__device__ __forceinline__ int kmap(int e, int half) {
    int r = e >> 1, j = e & 1;
    int base = (r < 4) ? (2 * r + 8 * half) : (16 + 2 * (r - 4) + 8 * half);
    return base + j;
}

__device__ __forceinline__ unsigned short f2bf(float f) {
    union { float f; unsigned u; } v; v.f = f;
    unsigned u = v.u;
    u += 0x7FFFu + ((u >> 16) & 1u);   // round-to-nearest-even
    return (unsigned short)(u >> 16);
}

// ---------------------------------------------------------------------------
// Pack a row-major f32 weight [N x ld] (used as B^T: out = x @ W^T) into
// bf16 B-fragments: dst[((nt*k_tiles + kt)*32 + lane)*16 + e]
// ---------------------------------------------------------------------------
__global__ void pack_b_kernel(const float* __restrict__ src,
                              unsigned short* __restrict__ dst,
                              int n_tiles, int k_tiles, int ld) {
    size_t total = (size_t)n_tiles * k_tiles * FRAG;
    for (size_t idx = blockIdx.x * (size_t)blockDim.x + threadIdx.x;
         idx < total; idx += (size_t)gridDim.x * blockDim.x) {
        int e    = (int)(idx & 15);
        int lane = (int)((idx >> 4) & 31);
        size_t kt = (idx >> 9) % (size_t)k_tiles;
        size_t nt = idx / ((size_t)FRAG * k_tiles);
        int n = (int)nt * 16 + (lane & 15);
        int k = (int)kt * 32 + kmap(e, lane >> 4);
        dst[idx] = f2bf(src[(size_t)n * ld + k]);
    }
}

// Pack f32 hidden [B x H] into bf16 A-fragments: dst[((kt*2+mt)*32+lane)*16+e]
__global__ void pack_a_kernel(const float* __restrict__ h,
                              unsigned short* __restrict__ dst) {
    int idx = blockIdx.x * blockDim.x + threadIdx.x;
    if (idx >= KT_H * 2 * FRAG) return;
    int e = idx & 15, lane = (idx >> 4) & 31, mt = (idx >> 9) & 1, kt = idx >> 10;
    int m = mt * 16 + (lane & 15);
    int k = kt * 32 + kmap(e, lane >> 4);
    dst[idx] = f2bf(h[m * H + k]);
}

// ---------------------------------------------------------------------------
// Attention (tiny; one block): scores over batch dim + context C[H]
// ---------------------------------------------------------------------------
__global__ void attn_kernel(const float* __restrict__ hidden0,  // [B,H] layer-0
                            const float* __restrict__ enc_h,    // [B,H]
                            const float* __restrict__ l1w, const float* __restrict__ l1b,
                            const float* __restrict__ l2w, const float* __restrict__ l2b,
                            float* __restrict__ C_ws,
                            float* __restrict__ scores_out, int t) {
    __shared__ float s_red[256];
    __shared__ float s_u[3];
    __shared__ float s_v[96];
    __shared__ float s_sc[32];
    int tid = threadIdx.x;
    // u_j = dot(l1w[j][0:H], hidden0[B-1])  (the broadcast 'a' vector)
    for (int j = 0; j < 3; ++j) {
        float p = 0.f;
        for (int h = tid; h < H; h += 256) p += l1w[j * K0 + h] * hidden0[(B - 1) * H + h];
        s_red[tid] = p; __syncthreads();
        for (int s = 128; s > 0; s >>= 1) { if (tid < s) s_red[tid] += s_red[tid + s]; __syncthreads(); }
        if (tid == 0) s_u[j] = s_red[0];
        __syncthreads();
    }
    // v_{b,j} = dot(l1w[j][H:2H], enc_h[b]) : one warp per pair
    int warp = tid >> 5, lane = tid & 31;
    for (int p = warp; p < 96; p += 8) {
        int b = p / 3, j = p % 3;
        float acc = 0.f;
        for (int h = lane; h < H; h += 32) acc += l1w[j * K0 + H + h] * enc_h[b * H + h];
        for (int m = 16; m > 0; m >>= 1) acc += __shfl_xor(acc, m, 32);
        if (lane == 0) s_v[p] = acc;
    }
    __syncthreads();
    // softmax over batch dim in warp 0
    if (warp == 0) {
        int b = lane;
        float s = l2b[0];
        for (int j = 0; j < 3; ++j) s += l2w[j] * (s_u[j] + s_v[b * 3 + j] + l1b[j]);
        float mx = s;
        for (int m = 16; m > 0; m >>= 1) mx = fmaxf(mx, __shfl_xor(mx, m, 32));
        float e = __expf(s - mx);
        float sum = e;
        for (int m = 16; m > 0; m >>= 1) sum += __shfl_xor(sum, m, 32);
        float sc = e / sum;
        s_sc[b] = sc;
        if (t == T - 1) scores_out[b] = sc;   // reference returns scores[-1]
    }
    __syncthreads();
    for (int h = tid; h < H; h += 256) {
        float c = 0.f;
        #pragma unroll
        for (int b = 0; b < B; ++b) c += enc_h[b * H + h] * s_sc[b];
        C_ws[h] = c;
    }
}

// Build layer-0 input x = [relu(emb[tok]) | C] directly as bf16 A-fragments
__global__ void pack_x_kernel(const float* __restrict__ emb,
                              const int* __restrict__ target,
                              const float* __restrict__ C_ws,
                              unsigned short* __restrict__ dst, int t) {
    int idx = blockIdx.x * blockDim.x + threadIdx.x;
    if (idx >= KT_X0 * 2 * FRAG) return;
    int e = idx & 15, lane = (idx >> 4) & 31, mt = (idx >> 9) & 1, kt = idx >> 10;
    int m = mt * 16 + (lane & 15);
    int k = kt * 32 + kmap(e, lane >> 4);
    float v;
    if (k < H) {
        int tok = (t == 0) ? 0 : target[m * T + (t - 1)];   // teacher forcing
        v = fmaxf(emb[(size_t)tok * H + k], 0.f);
    } else {
        v = C_ws[k - H];
    }
    dst[idx] = f2bf(v);
}

// ---------------------------------------------------------------------------
// GRU gates: gi = x @ w_ih^T, gh = h @ w_hh^T  (bf16 WMMA, f32 accumulate)
// One wave covers a FULL 32-row x 16-col output stripe (both m-tiles), so
// each B fragment is loaded exactly once: 192 waves -> 24 blocks x 8 waves.
// ---------------------------------------------------------------------------
__global__ void gates_kernel(const unsigned short* __restrict__ afragX, int ktX,
                             const unsigned short* __restrict__ bfragI,
                             const unsigned short* __restrict__ afragH,
                             const unsigned short* __restrict__ bfragH,
                             float* __restrict__ gi, float* __restrict__ gh) {
    int nt   = (blockIdx.x * blockDim.x + threadIdx.x) >> 5;
    int lane = threadIdx.x & 31;
    if (nt >= NT_G) return;

    v8f acc_i0 = {}, acc_i1 = {};
    for (int kt = 0; kt < ktX; ++kt) {
        v16bf b  = *reinterpret_cast<const v16bf*>(bfragI + ((size_t)nt * ktX + kt) * FRAG + lane * 16);
        v16bf a0 = *reinterpret_cast<const v16bf*>(afragX + ((size_t)(kt * 2 + 0)) * FRAG + lane * 16);
        v16bf a1 = *reinterpret_cast<const v16bf*>(afragX + ((size_t)(kt * 2 + 1)) * FRAG + lane * 16);
        acc_i0 = __builtin_amdgcn_wmma_f32_16x16x32_bf16(false, a0, false, b, (short)0, acc_i0, false, false);
        acc_i1 = __builtin_amdgcn_wmma_f32_16x16x32_bf16(false, a1, false, b, (short)0, acc_i1, false, false);
    }
    v8f acc_h0 = {}, acc_h1 = {};
    for (int kt = 0; kt < KT_H; ++kt) {
        v16bf b  = *reinterpret_cast<const v16bf*>(bfragH + ((size_t)nt * KT_H + kt) * FRAG + lane * 16);
        v16bf a0 = *reinterpret_cast<const v16bf*>(afragH + ((size_t)(kt * 2 + 0)) * FRAG + lane * 16);
        v16bf a1 = *reinterpret_cast<const v16bf*>(afragH + ((size_t)(kt * 2 + 1)) * FRAG + lane * 16);
        acc_h0 = __builtin_amdgcn_wmma_f32_16x16x32_bf16(false, a0, false, b, (short)0, acc_h0, false, false);
        acc_h1 = __builtin_amdgcn_wmma_f32_16x16x32_bf16(false, a1, false, b, (short)0, acc_h1, false, false);
    }
    // D layout: lane 0-15 -> n=lane, m=r ; lane 16-31 -> n=lane-16, m=8+r
    int n = nt * 16 + (lane & 15);
    int mbase = (lane >> 4) << 3;
    #pragma unroll
    for (int r = 0; r < 8; ++r) {
        gi[(size_t)(mbase + r) * H3 + n]      = acc_i0[r];
        gi[(size_t)(16 + mbase + r) * H3 + n] = acc_i1[r];
        gh[(size_t)(mbase + r) * H3 + n]      = acc_h0[r];
        gh[(size_t)(16 + mbase + r) * H3 + n] = acc_h1[r];
    }
}

// GRU pointwise: h' = (1-z)*n + z*h ; also re-pack h' into A-fragments
__global__ void gru_pw_kernel(const float* __restrict__ gi, const float* __restrict__ gh,
                              const float* __restrict__ b_ih, const float* __restrict__ b_hh,
                              float* __restrict__ hcur, unsigned short* __restrict__ afragH) {
    int idx = blockIdx.x * blockDim.x + threadIdx.x;
    if (idx >= B * H) return;
    int b = idx / H, j = idx % H;
    size_t r0 = (size_t)b * H3;
    float ir  = gi[r0 + j]         + b_ih[j];
    float iz  = gi[r0 + H + j]     + b_ih[H + j];
    float in_ = gi[r0 + 2 * H + j] + b_ih[2 * H + j];
    float hr  = gh[r0 + j]         + b_hh[j];
    float hz  = gh[r0 + H + j]     + b_hh[H + j];
    float hn  = gh[r0 + 2 * H + j] + b_hh[2 * H + j];
    float rg = 1.f / (1.f + __expf(-(ir + hr)));
    float zg = 1.f / (1.f + __expf(-(iz + hz)));
    float ng = tanhf(in_ + rg * hn);
    float hprev = hcur[idx];
    float hnew = (1.f - zg) * ng + zg * hprev;
    hcur[idx] = hnew;
    // inverse kmap: place hnew into A-fragment slot
    int kt = j >> 5, kl = j & 31;
    int half = (kl >> 3) & 1;
    int e = (kl < 16) ? (kl & 7) : (8 + (kl & 7));
    int mt = b >> 4, lane = (half << 4) | (b & 15);
    afragH[((size_t)(kt * 2 + mt)) * FRAG + lane * 16 + e] = f2bf(hnew);
}

// ---------------------------------------------------------------------------
// Vocab projection: logits[b, t, :] = h3 @ out_w^T + out_b
// One wave per 16-wide N stripe, both m-tiles -> each out_w fragment read
// once per step: 2000 waves -> 250 blocks x 8 waves.
// ---------------------------------------------------------------------------
__global__ void logits_kernel(const unsigned short* __restrict__ afragH,
                              const unsigned short* __restrict__ bfragW,
                              const float* __restrict__ out_b,
                              float* __restrict__ out, int t) {
    int nt   = (blockIdx.x * blockDim.x + threadIdx.x) >> 5;
    int lane = threadIdx.x & 31;
    if (nt >= NT_V) return;
    const unsigned short* bp = bfragW + (size_t)nt * KT_H * FRAG;
    v8f acc0 = {}, acc1 = {};
    for (int kt = 0; kt < KT_H; ++kt) {
        if (kt + 4 < KT_H)
            __builtin_prefetch(bp + (size_t)(kt + 4) * FRAG + lane * 16, 0, 1);
        v16bf b  = *reinterpret_cast<const v16bf*>(bp + (size_t)kt * FRAG + lane * 16);
        v16bf a0 = *reinterpret_cast<const v16bf*>(afragH + ((size_t)(kt * 2 + 0)) * FRAG + lane * 16);
        v16bf a1 = *reinterpret_cast<const v16bf*>(afragH + ((size_t)(kt * 2 + 1)) * FRAG + lane * 16);
        acc0 = __builtin_amdgcn_wmma_f32_16x16x32_bf16(false, a0, false, b, (short)0, acc0, false, false);
        acc1 = __builtin_amdgcn_wmma_f32_16x16x32_bf16(false, a1, false, b, (short)0, acc1, false, false);
    }
    int n = nt * 16 + (lane & 15);
    float bias = out_b[n];
    int mbase = (lane >> 4) << 3;
    #pragma unroll
    for (int r = 0; r < 8; ++r) {
        out[((size_t)(mbase + r) * T + t) * V + n]      = acc0[r] + bias;
        out[((size_t)(16 + mbase + r) * T + t) * V + n] = acc1[r] + bias;
    }
}

// In-place log-softmax over V per (b,t) row
__global__ void logsoftmax_kernel(float* __restrict__ out) {
    __shared__ float s_red[256];
    size_t base = (size_t)blockIdx.x * V;
    int tid = threadIdx.x;
    float mx = -3.4e38f;
    for (int v = tid; v < V; v += 256) mx = fmaxf(mx, out[base + v]);
    s_red[tid] = mx; __syncthreads();
    for (int s = 128; s > 0; s >>= 1) { if (tid < s) s_red[tid] = fmaxf(s_red[tid], s_red[tid + s]); __syncthreads(); }
    mx = s_red[0]; __syncthreads();
    float sum = 0.f;
    for (int v = tid; v < V; v += 256) sum += __expf(out[base + v] - mx);
    s_red[tid] = sum; __syncthreads();
    for (int s = 128; s > 0; s >>= 1) { if (tid < s) s_red[tid] += s_red[tid + s]; __syncthreads(); }
    float lse = mx + __logf(s_red[0]);
    __syncthreads();
    for (int v = tid; v < V; v += 256) out[base + v] -= lse;
}

// ---------------------------------------------------------------------------
// Host orchestration
// ---------------------------------------------------------------------------
extern "C" void kernel_launch(void* const* d_in, const int* in_sizes, int n_in,
                              void* d_out, int out_size, void* d_ws, size_t ws_size,
                              hipStream_t stream) {
    (void)in_sizes; (void)n_in; (void)out_size; (void)ws_size; (void)S;
    const float* enc_hid = (const float*)d_in[1];   // [L,B,H]; enc_h = first B*H
    const int*   target  = (const int*)d_in[2];
    const float* emb     = (const float*)d_in[3];
    const float* w_ih0   = (const float*)d_in[4];
    const float* w_hh0   = (const float*)d_in[5];
    const float* b_ih0   = (const float*)d_in[6];
    const float* b_hh0   = (const float*)d_in[7];
    const float* w_ih    = (const float*)d_in[8];
    const float* w_hh    = (const float*)d_in[9];
    const float* b_ih    = (const float*)d_in[10];
    const float* b_hh    = (const float*)d_in[11];
    const float* out_w   = (const float*)d_in[12];
    const float* out_b   = (const float*)d_in[13];
    const float* l1w     = (const float*)d_in[14];
    const float* l1b     = (const float*)d_in[15];
    const float* l2w     = (const float*)d_in[16];
    const float* l2b     = (const float*)d_in[17];
    float* out = (float*)d_out;

    char* ws = (char*)d_ws;
    size_t off = 0;
    auto give = [&](size_t bytes) -> char* {
        char* p = ws + off;
        off = (off + bytes + 255) & ~(size_t)255;
        return p;
    };
    unsigned short* fOutW = (unsigned short*)give((size_t)NT_V * KT_H * FRAG * 2);
    unsigned short* fWih0 = (unsigned short*)give((size_t)NT_G * KT_X0 * FRAG * 2);
    unsigned short* fWhh0 = (unsigned short*)give((size_t)NT_G * KT_H * FRAG * 2);
    unsigned short *fWih[3], *fWhh[3];
    for (int l = 0; l < 3; ++l) {
        fWih[l] = (unsigned short*)give((size_t)NT_G * KT_H * FRAG * 2);
        fWhh[l] = (unsigned short*)give((size_t)NT_G * KT_H * FRAG * 2);
    }
    unsigned short* fX = (unsigned short*)give((size_t)KT_X0 * 2 * FRAG * 2);
    unsigned short* fH[4];
    for (int l = 0; l < 4; ++l) fH[l] = (unsigned short*)give((size_t)KT_H * 2 * FRAG * 2);
    float* hidden = (float*)give((size_t)L * B * H * 4);
    float* gi     = (float*)give((size_t)B * H3 * 4);
    float* gh     = (float*)give((size_t)B * H3 * 4);
    float* Cv     = (float*)give((size_t)H * 4);

    // One-time weight packing (bf16 fragment layout; total ~122 MB -> L2-resident)
    pack_b_kernel<<<4096, 256, 0, stream>>>(out_w, fOutW, NT_V, KT_H, H);
    pack_b_kernel<<<1024, 256, 0, stream>>>(w_ih0, fWih0, NT_G, KT_X0, K0);
    pack_b_kernel<<<512, 256, 0, stream>>>(w_hh0, fWhh0, NT_G, KT_H, H);
    for (int l = 0; l < 3; ++l) {
        pack_b_kernel<<<512, 256, 0, stream>>>(w_ih + (size_t)l * H3 * H, fWih[l], NT_G, KT_H, H);
        pack_b_kernel<<<512, 256, 0, stream>>>(w_hh + (size_t)l * H3 * H, fWhh[l], NT_G, KT_H, H);
    }
    hipMemcpyAsync(hidden, enc_hid, (size_t)L * B * H * 4, hipMemcpyDeviceToDevice, stream);
    for (int l = 0; l < 4; ++l)
        pack_a_kernel<<<128, 256, 0, stream>>>(hidden + (size_t)l * B * H, fH[l]);

    float* scores_out = out + (size_t)B * T * V + (size_t)L * B * H;
    for (int t = 0; t < T; ++t) {
        attn_kernel<<<1, 256, 0, stream>>>(hidden, enc_hid, l1w, l1b, l2w, l2b, Cv, scores_out, t);
        pack_x_kernel<<<256, 256, 0, stream>>>(emb, target, Cv, fX, t);
        gates_kernel<<<24, 256, 0, stream>>>(fX, KT_X0, fWih0, fH[0], fWhh0, gi, gh);
        gru_pw_kernel<<<128, 256, 0, stream>>>(gi, gh, b_ih0, b_hh0, hidden, fH[0]);
        for (int l = 1; l < 4; ++l) {
            gates_kernel<<<24, 256, 0, stream>>>(fH[l - 1], KT_H, fWih[l - 1], fH[l], fWhh[l - 1], gi, gh);
            gru_pw_kernel<<<128, 256, 0, stream>>>(gi, gh,
                                                   b_ih + (size_t)(l - 1) * H3,
                                                   b_hh + (size_t)(l - 1) * H3,
                                                   hidden + (size_t)l * B * H, fH[l]);
        }
        logits_kernel<<<250, 256, 0, stream>>>(fH[3], fOutW, out_b, out, t);
    }
    logsoftmax_kernel<<<B * T, 256, 0, stream>>>(out);
    hipMemcpyAsync(out + (size_t)B * T * V, hidden, (size_t)L * B * H * 4,
                   hipMemcpyDeviceToDevice, stream);
}
